// Quantizer_10041633538950
// MI455X (gfx1250) — matrix operand order repaired and checked by reading, not compiled, not run on hardware
//
#include <hip/hip_runtime.h>
#include <hip/hip_bf16.h>
#include <math.h>

typedef __attribute__((ext_vector_type(16))) __bf16 v16bf;
typedef __attribute__((ext_vector_type(8)))  float  v8f;

#define NB 16
#define DC 64
#define TL 4096
#define NROWS (NB*TL)        // 65536
#define KCODES 1024
#define NSTAGE 8
#define ROWS_PER_BLK 128
#define NBLK (NROWS/ROWS_PER_BLK)  // 512
#define RPAD 68              // padded row stride (floats) in LDS residual tile

// ---------------- codebook prep: f32 -> bf16 + per-code squared norms ----------------
__global__ void prep_cb(const float* __restrict__ cb, __bf16* __restrict__ cbbf,
                        float* __restrict__ ck2) {
    int r = blockIdx.x * blockDim.x + threadIdx.x;   // 0..8191 code rows
    if (r >= NSTAGE * KCODES) return;
    const float* src = cb + (size_t)r * DC;
    __bf16* dst = cbbf + (size_t)r * DC;
    float s = 0.f;
    for (int d = 0; d < DC; ++d) {
        float v = src[d];
        s += v * v;
        dst[d] = (__bf16)v;
    }
    ck2[r] = s;
}

// ---------------- z [B,D,T] -> residual [B*T, D] (tiled LDS transpose) ----------------
__global__ void transpose_in(const float* __restrict__ z, float* __restrict__ resid) {
    __shared__ float tile[64 * 65];
    int b  = blockIdx.x >> 6;
    int t0 = (blockIdx.x & 63) << 6;
    for (int i = 0; i < 16; ++i) {
        int lin = threadIdx.x + i * 256;
        int d = lin >> 6, t = lin & 63;
        tile[d * 65 + t] = z[((size_t)(b * 64 + d)) * TL + t0 + t];
    }
    __syncthreads();
    for (int i = 0; i < 16; ++i) {
        int lin = threadIdx.x + i * 256;
        int t = lin >> 6, d = lin & 63;
        resid[((size_t)(b * TL + t0 + t)) * DC + d] = tile[d * 65 + t];
    }
}

// ---------------- fused stage: WMMA argmin + residual update + loss + histogram --------
__global__ __launch_bounds__(256) void rvq_stage(
    float* __restrict__ resid,
    const __bf16* __restrict__ cbbf,
    const float* __restrict__ cbf32,
    const float* __restrict__ ck2,
    unsigned int* __restrict__ counts,
    float* __restrict__ partials,
    int stage) {
    __shared__ float rtile[ROWS_PER_BLK * RPAD];   // padded residual tile (f32)
    __shared__ int   idxArr[ROWS_PER_BLK];
    __shared__ float wsum[8];

    const int t = threadIdx.x;
    const int rowbase = blockIdx.x * ROWS_PER_BLK;
    const __bf16* cbs   = cbbf  + (size_t)stage * KCODES * DC;
    const float*  cbs32 = cbf32 + (size_t)stage * KCODES * DC;
    const float*  ck2s  = ck2   + stage * KCODES;

    // phase 1: load 128x64 f32 residual tile into padded LDS (coalesced float4)
    {
        const float4* g = (const float4*)(resid + (size_t)rowbase * DC);
        for (int i = 0; i < 8; ++i) {
            int lin4 = t + i * 256;              // 0..2047
            int row = lin4 >> 4, c4 = lin4 & 15;
            float4 v = g[lin4];
            *(float4*)&rtile[row * RPAD + c4 * 4] = v;
        }
    }
    __syncthreads();

    // phase 2: each wave -> 16 rows x 1024 codes, v_wmma_f32_16x16x32_bf16
    {
        const int wave = t >> 5, lane = t & 31;
        const int wr0 = wave * 16;
        const int l15 = lane & 15;
        const int arow = wr0 + l15;
        const int kb = (lane < 16) ? 0 : 16;     // A layout: K range [kb, kb+16)
        const float* rr = &rtile[arow * RPAD];
        v16bf a0, a1;
        #pragma unroll
        for (int e = 0; e < 8; ++e) {
            a0[e]     = (__bf16)rr[kb + e];
            a0[8 + e] = (__bf16)rr[kb + 8 + e];
            a1[e]     = (__bf16)rr[32 + kb + e];
            a1[8 + e] = (__bf16)rr[32 + kb + 8 + e];
        }
        float bval[8]; int bidx[8];
        #pragma unroll
        for (int v = 0; v < 8; ++v) { bval[v] = 3.4e38f; bidx[v] = 0; }

        const int kb2 = (lane < 16) ? 0 : 16;    // B layout: K range [kb2, kb2+16)
        #pragma unroll 2
        for (int ct = 0; ct < KCODES / 16; ++ct) {
            int n = ct * 16 + l15;               // this lane's code column
            const __bf16* cn = cbs + n * DC;
            v16bf b0 = *(const v16bf*)(cn + kb2);        // K 0..31 slice
            v16bf b1 = *(const v16bf*)(cn + 32 + kb2);   // K 32..63 slice
            float c2 = ck2s[n];
            v8f acc = {};
            acc = __builtin_amdgcn_wmma_f32_16x16x32_bf16(false, a0, false, b0,
                                                          (short)0, acc, false, false);
            acc = __builtin_amdgcn_wmma_f32_16x16x32_bf16(false, a1, false, b1,
                                                          (short)0, acc, false, false);
            #pragma unroll
            for (int v = 0; v < 8; ++v) {
                float m = c2 - 2.0f * acc[v];    // ||c||^2 - 2 x.c
                if (m < bval[v]) { bval[v] = m; bidx[v] = n; }
            }
        }
        // reduce (val,idx) across the 16 lanes of each half; lower idx wins ties
        #pragma unroll
        for (int v = 0; v < 8; ++v) {
            float val = bval[v]; int idx = bidx[v];
            #pragma unroll
            for (int m = 8; m >= 1; m >>= 1) {
                float ov = __shfl_xor(val, m, 32);
                int   oi = __shfl_xor(idx, m, 32);
                if (ov < val || (ov == val && oi < idx)) { val = ov; idx = oi; }
            }
            bval[v] = val; bidx[v] = idx;
        }
        if (l15 == 0) {
            int rb = wr0 + ((lane >> 4) << 3);   // lane 0 -> rows v, lane 16 -> rows v+8
            #pragma unroll
            for (int v = 0; v < 8; ++v) idxArr[rb + v] = bidx[v];
        }
    }
    __syncthreads();

    // phase 3: gather q (f32), update residual, accumulate loss, histogram
    float sumsq = 0.f;
    {
        float4* g = (float4*)(resid + (size_t)rowbase * DC);
        for (int i = 0; i < 8; ++i) {
            int lin4 = t + i * 256;
            int row = lin4 >> 4, c4 = lin4 & 15;
            int ci = idxArr[row];
            float4 q = *(const float4*)(cbs32 + (size_t)ci * DC + c4 * 4);
            float4 r = *(const float4*)&rtile[row * RPAD + c4 * 4];
            float4 nr;
            nr.x = r.x - q.x; nr.y = r.y - q.y; nr.z = r.z - q.z; nr.w = r.w - q.w;
            g[lin4] = nr;
            sumsq += nr.x*nr.x + nr.y*nr.y + nr.z*nr.z + nr.w*nr.w;
        }
    }
    if (t < ROWS_PER_BLK)
        atomicAdd(&counts[stage * KCODES + idxArr[t]], 1u);

    // deterministic block reduction of sumsq
    #pragma unroll
    for (int m = 16; m >= 1; m >>= 1) sumsq += __shfl_xor(sumsq, m, 32);
    if ((t & 31) == 0) wsum[t >> 5] = sumsq;
    __syncthreads();
    if (t == 0) {
        float tot = 0.f;
        for (int w = 0; w < 8; ++w) tot += wsum[w];
        partials[stage * NBLK + blockIdx.x] = tot;
    }
}

// ---------------- zq = z - residual (tiled transpose back to [B,D,T]) ----------------
__global__ void finalize_out(const float* __restrict__ z, const float* __restrict__ resid,
                             float* __restrict__ out) {
    __shared__ float tile[64 * 65];
    int b  = blockIdx.x >> 6;
    int t0 = (blockIdx.x & 63) << 6;
    for (int i = 0; i < 16; ++i) {
        int lin = threadIdx.x + i * 256;
        int r = lin >> 6, d = lin & 63;
        tile[r * 65 + d] = resid[((size_t)(b * TL + t0 + r)) * DC + d];
    }
    __syncthreads();
    for (int i = 0; i < 16; ++i) {
        int lin = threadIdx.x + i * 256;
        int d = lin >> 6, tt = lin & 63;
        size_t gi = ((size_t)(b * 64 + d)) * TL + t0 + tt;
        out[gi] = z[gi] - tile[tt * 65 + d];
    }
}

// ---------------- losses + perplexities (fixed-order, deterministic) ----------------
__global__ void finalize_stats(const float* __restrict__ partials,
                               const unsigned int* __restrict__ counts,
                               float* __restrict__ out_tail) {
    int s = threadIdx.x;
    if (s < NSTAGE) {
        float tot = 0.f;
        for (int i = 0; i < NBLK; ++i) tot += partials[s * NBLK + i];
        out_tail[s] = tot / (float)((size_t)NROWS * DC);
    } else if (s >= 32 && s < 32 + NSTAGE) {
        int ss = s - 32;
        float H = 0.f;
        for (int k = 0; k < KCODES; ++k) {
            float p = (float)counts[ss * KCODES + k] / (float)NROWS;
            H += p * logf(p + 1e-10f);
        }
        out_tail[NSTAGE + ss] = expf(-H);
    }
}

extern "C" void kernel_launch(void* const* d_in, const int* in_sizes, int n_in,
                              void* d_out, int out_size, void* d_ws, size_t ws_size,
                              hipStream_t stream) {
    const float* z  = (const float*)d_in[0];   // [16, 64, 4096]
    const float* cb = (const float*)d_in[1];   // [8, 1024, 64]
    float* out = (float*)d_out;                // [16*64*4096] ++ losses[8] ++ perp[8]
    char* ws = (char*)d_ws;

    float*        resid    = (float*)ws;                               // 16 MB
    __bf16*       cbbf     = (__bf16*)(ws + 16777216);                 // 1 MB
    float*        ck2      = (float*)(ws + 17825792);                  // 32 KB
    unsigned int* counts   = (unsigned int*)(ws + 17858560);           // 32 KB
    float*        partials = (float*)(ws + 17891328);                  // 16 KB

    hipMemsetAsync(counts, 0, 32768 + 16384, stream);  // counts + partials
    prep_cb<<<(NSTAGE * KCODES + 255) / 256, 256, 0, stream>>>(cb, cbbf, ck2);
    transpose_in<<<NB * (TL / 64), 256, 0, stream>>>(z, resid);
    for (int s = 0; s < NSTAGE; ++s)
        rvq_stage<<<NBLK, 256, 0, stream>>>(resid, cbbf, cb, ck2, counts, partials, s);
    finalize_out<<<NB * (TL / 64), 256, 0, stream>>>(z, resid, out);
    finalize_stats<<<1, 64, 0, stream>>>(partials, counts, out + (size_t)NB * DC * TL);
}
